// SingleGCNEncoder_89850715832383
// MI455X (gfx1250) — compile-verified, matrix operand order checked by reading, not run on hardware
//
#include <hip/hip_runtime.h>
#include <hip/hip_bf16.h>

// ---------------------------------------------------------------------------
// 2-layer GCN encoder for MI455X (gfx1250, wave32).
// GEMMs via v_wmma_f32_16x16x32_bf16 with fragment-major LDS staging
// (2x ds_load_b128 per operand, bank-conflict-free padded stride).
// Edge scatter via float4 gathers + global_atomic_add_f32
// (working set ~102MB -> resolves in 192MB L2; edge stream from 23.3TB/s HBM).
// ---------------------------------------------------------------------------

typedef __attribute__((ext_vector_type(16))) __bf16 v16bf;
typedef __attribute__((ext_vector_type(8)))  __bf16 v8bf;
typedef __attribute__((ext_vector_type(2)))  __bf16 v2bf;
typedef __attribute__((ext_vector_type(8)))  float  v8f;

__device__ __forceinline__ void store_bf16_pair(__bf16* p, float a, float b) {
  v2bf v; v[0] = (__bf16)a; v[1] = (__bf16)b;
  *reinterpret_cast<v2bf*>(p) = v;
}

// K-permutation of the 16-bit WMMA fragment (ISA 7.12.2, 16x32 A / 32x16 B):
//   lane-half g = (k>>3)&1 ; element j = (k&7) + 8*((k>>4)&1)
// Even/odd k pairs keep same (g, blk) and consecutive j -> dword-packable.

// ---------------- degree / normalization ----------------

__global__ void fill_ones(float* __restrict__ deg, long long n) {
  long long t = (long long)blockIdx.x * blockDim.x + threadIdx.x;
  if (t < n) deg[t] = 1.0f;  // self loop
}

__global__ void deg_count(const long long* __restrict__ dst, float* __restrict__ deg,
                          long long E) {
  long long t = (long long)blockIdx.x * blockDim.x + threadIdx.x;
  if (t < E) atomicAdd(&deg[dst[t]], 1.0f);
}

__global__ void rsqrt_inplace(float* __restrict__ d, long long n) {
  long long t = (long long)blockIdx.x * blockDim.x + threadIdx.x;
  if (t < n) d[t] = rsqrtf(d[t]);
}

// ---------------- WMMA bf16 GEMM: H[N,64] = X[N,K] @ W[K,64] ----------------
// 256 threads = 8 waves; each wave: one 16x16 f32 tile; block: 32 rows x 64 cols.

template <int K>
__global__ __launch_bounds__(256) void gemm_wmma_bf16(
    const float* __restrict__ X, const float* __restrict__ W,
    float* __restrict__ H, int Nrows) {
  constexpr int NB = K / 32;            // k-blocks per row
  constexpr int RS = NB * 32 + 8;       // padded fragment-row stride (elements)
  __shared__ __attribute__((aligned(16))) __bf16 Xp[32 * RS];  // [row][blk][g][j]
  __shared__ __attribute__((aligned(16))) __bf16 Wp[64 * RS];  // [col][blk][g][j]

  const int tid = threadIdx.x;
  const int rowBase = blockIdx.x * 32;
  const bool fullTile = (rowBase + 32 <= Nrows);

  // ---- stage X tile: float2 global loads, permuted packed-bf16 LDS stores ----
  const float2* X2 = reinterpret_cast<const float2*>(X);
  for (int p = tid; p < 32 * (K / 2); p += 256) {
    int r  = p / (K / 2);
    int kp = p - r * (K / 2);
    int k0 = kp * 2;
    int gr = rowBase + r;
    float2 v = make_float2(0.f, 0.f);
    if (fullTile || gr < Nrows) v = X2[(size_t)gr * (K / 2) + kp];
    int kl  = k0 & 31;
    int blk = k0 >> 5;
    int g   = (kl >> 3) & 1;
    int j   = (kl & 7) + 8 * ((kl >> 4) & 1);
    store_bf16_pair(&Xp[r * RS + blk * 32 + g * 16 + j], v.x, v.y);
  }
  // ---- stage W (K x 64 row-major): two strided scalar loads per packed pair ----
  for (int p = tid; p < (K / 2) * 64; p += 256) {
    int kp  = p >> 6;
    int col = p & 63;
    int k0  = kp * 2;
    float w0 = W[(size_t)k0 * 64 + col];
    float w1 = W[(size_t)(k0 + 1) * 64 + col];
    int kl  = k0 & 31;
    int blk = k0 >> 5;
    int g   = (kl >> 3) & 1;
    int j   = (kl & 7) + 8 * ((kl >> 4) & 1);
    store_bf16_pair(&Wp[col * RS + blk * 32 + g * 16 + j], w0, w1);
  }
  __syncthreads();

  const int wave  = tid >> 5;
  const int lane  = tid & 31;
  const int rtile = wave >> 2;   // 0..1
  const int ctile = wave & 3;    // 0..3
  const int g     = lane >> 4;
  const int m     = lane & 15;

  const __bf16* ap = &Xp[(rtile * 16 + m) * RS + g * 16];
  const __bf16* bp = &Wp[(ctile * 16 + m) * RS + g * 16];

  v8f acc = {};
#pragma unroll
  for (int blk = 0; blk < NB; ++blk) {
    v8bf alo = *reinterpret_cast<const v8bf*>(ap + blk * 32);
    v8bf ahi = *reinterpret_cast<const v8bf*>(ap + blk * 32 + 8);
    v8bf blo = *reinterpret_cast<const v8bf*>(bp + blk * 32);
    v8bf bhi = *reinterpret_cast<const v8bf*>(bp + blk * 32 + 8);
    v16bf a = __builtin_shufflevector(alo, ahi, 0,1,2,3,4,5,6,7,8,9,10,11,12,13,14,15);
    v16bf b = __builtin_shufflevector(blo, bhi, 0,1,2,3,4,5,6,7,8,9,10,11,12,13,14,15);
    acc = __builtin_amdgcn_wmma_f32_16x16x32_bf16(
        /*neg_a=*/false, a, /*neg_b=*/false, b,
        /*c_mod=*/(short)0, acc, /*reuse_a=*/false, /*reuse_b=*/false);
  }

  // C/D layout: VGPR v -> M = v + 8*laneHalf, N = m (ISA 7.12.2)
  const int row0 = rowBase + rtile * 16 + 8 * g;     // first of 8 output rows
  float* Hb = H + (size_t)row0 * 64 + ctile * 16 + m;
  if (fullTile) {
#pragma unroll
    for (int v = 0; v < 8; ++v) Hb[(size_t)v * 64] = acc[v];  // immediate offsets
  } else {
#pragma unroll
    for (int v = 0; v < 8; ++v)
      if (row0 + v < Nrows) Hb[(size_t)v * 64] = acc[v];
  }
}

// ---------------- aggregation ----------------

// agg[n][0:64] = h[n][0:64] * dinv[n]^2   (self-loop term doubles as init)
__global__ void selfloop_init(const float4* __restrict__ h4, const float* __restrict__ dinv,
                              float4* __restrict__ agg4, long long N) {
  long long t = (long long)blockIdx.x * blockDim.x + threadIdx.x;
  if (t >= N * 16) return;
  long long n = t >> 4;
  float di = dinv[n];
  float w = di * di;
  float4 v = h4[t];
  v.x *= w; v.y *= w; v.z *= w; v.w *= w;
  agg4[t] = v;
}

// agg[dst][f:f+4] += h[src][f:f+4] * dinv[src]*dinv[dst]; one thread per (edge, f4)
__global__ void scatter_edges(const long long* __restrict__ src,
                              const long long* __restrict__ dst,
                              const float* __restrict__ dinv,
                              const float* __restrict__ h,
                              float* __restrict__ agg, long long E) {
  long long t = (long long)blockIdx.x * blockDim.x + threadIdx.x;
  if (t >= E * 16) return;
  long long e = t >> 4;
  int f = (int)(t & 15) * 4;
  long long s = src[e], d = dst[e];
  float w = dinv[s] * dinv[d];
  float4 v = *reinterpret_cast<const float4*>(&h[s * 64 + f]);
  float* ad = &agg[d * 64 + f];
  atomicAdd(ad + 0, v.x * w);
  atomicAdd(ad + 1, v.y * w);
  atomicAdd(ad + 2, v.z * w);
  atomicAdd(ad + 3, v.w * w);
}

template <bool RELU>
__global__ void bias_act(float4* __restrict__ a, const float4* __restrict__ b, long long N) {
  long long t = (long long)blockIdx.x * blockDim.x + threadIdx.x;
  if (t >= N * 16) return;
  float4 v = a[t];
  float4 bb = b[t & 15];
  v.x += bb.x; v.y += bb.y; v.z += bb.z; v.w += bb.w;
  if (RELU) {
    v.x = fmaxf(v.x, 0.f); v.y = fmaxf(v.y, 0.f);
    v.z = fmaxf(v.z, 0.f); v.w = fmaxf(v.w, 0.f);
  }
  a[t] = v;
}

// ---------------- launch ----------------

extern "C" void kernel_launch(void* const* d_in, const int* in_sizes, int n_in,
                              void* d_out, int out_size, void* d_ws, size_t ws_size,
                              hipStream_t stream) {
  const float*     x  = (const float*)d_in[0];       // [N,128]
  const long long* ei = (const long long*)d_in[1];   // [2,E] int64
  const float*     W1 = (const float*)d_in[2];       // [128,64]
  const float*     b1 = (const float*)d_in[3];       // [64]
  const float*     W2 = (const float*)d_in[4];       // [64,64]
  const float*     b2 = (const float*)d_in[5];       // [64]
  float* out = (float*)d_out;                        // [N,64]

  const long long N = in_sizes[0] / 128;
  const long long E = in_sizes[1] / 2;
  const long long* src = ei;
  const long long* dst = ei + E;

  float* ws   = (float*)d_ws;
  size_t oD   = ((size_t)N + 255) & ~(size_t)255;
  float* dinv = ws;                       // [N]
  float* h    = ws + oD;                  // [N,64]  (layer1 h, then layer2 h)
  float* agg  = h + (size_t)N * 64;       // [N,64]  (layer1 agg / z1)

  const int B = 256;
  int gN  = (int)((N + B - 1) / B);
  int gE  = (int)((E + B - 1) / B);
  int gN4 = (int)((N * 16 + B - 1) / B);
  int gE4 = (int)((E * 16 + B - 1) / B);
  int gMM = (int)((N + 31) / 32);

  // normalization
  fill_ones<<<gN, B, 0, stream>>>(dinv, N);
  deg_count<<<gE, B, 0, stream>>>(dst, dinv, E);
  rsqrt_inplace<<<gN, B, 0, stream>>>(dinv, N);

  // ---- layer 1: h = x@W1 ; agg = selfloop + scatter ; z1 = relu(agg+b1) ----
  gemm_wmma_bf16<128><<<gMM, B, 0, stream>>>(x, W1, h, (int)N);
  selfloop_init<<<gN4, B, 0, stream>>>((const float4*)h, dinv, (float4*)agg, N);
  scatter_edges<<<gE4, B, 0, stream>>>(src, dst, dinv, h, agg, E);
  bias_act<true><<<gN4, B, 0, stream>>>((float4*)agg, (const float4*)b1, N);

  // ---- layer 2: h = z1@W2 ; out = selfloop + scatter + b2 ----
  gemm_wmma_bf16<64><<<gMM, B, 0, stream>>>(agg, W2, h, (int)N);
  selfloop_init<<<gN4, B, 0, stream>>>((const float4*)h, dinv, (float4*)out, N);
  scatter_edges<<<gE4, B, 0, stream>>>(src, dst, dinv, h, out, E);
  bias_act<false><<<gN4, B, 0, stream>>>((float4*)out, (const float4*)b2, N);
}
